// RetNetDecoderLayer_3161095930393
// MI455X (gfx1250) — matrix operand order repaired and checked by reading, not compiled, not run on hardware
//
#include <hip/hip_runtime.h>
#include <hip/hip_bf16.h>
#include <stdint.h>

// ---------------- types ----------------
typedef __attribute__((ext_vector_type(16))) __bf16 bf16x16;
typedef __attribute__((ext_vector_type(8)))  __bf16 bf16x8;
typedef __attribute__((ext_vector_type(8)))  float  f32x8;
typedef __attribute__((ext_vector_type(4)))  int    i32x4;

#if __has_builtin(__builtin_amdgcn_global_load_async_to_lds_b128)
#define HAVE_ASYNC_B128 1
#else
#define HAVE_ASYNC_B128 0
#endif
#if __has_builtin(__builtin_amdgcn_s_wait_asynccnt)
#define HAVE_WAIT_ASYNC 1
#else
#define HAVE_WAIT_ASYNC 0
#endif

#if HAVE_ASYNC_B128
typedef __attribute__((address_space(1))) i32x4 glob_i32x4;
typedef __attribute__((address_space(3))) i32x4 lds_i32x4;
#endif

// ---------------- scalar helpers ----------------
__device__ __forceinline__ __bf16 f2bf(float x) {
  union { unsigned int u; float f; } v; v.f = x;
  unsigned int u = v.u + 0x7FFFu + ((v.u >> 16) & 1u);   // round-to-nearest-even
  union { unsigned short u; __bf16 b; } r; r.u = (unsigned short)(u >> 16);
  return r.b;
}
__device__ __forceinline__ float gelu_exact(float x) {
  return 0.5f * x * (1.0f + erff(x * 0.70710678118654752f));
}

// 16B global -> LDS copy; async path on CDNA5.
__device__ __forceinline__ void copy16_g2lds(const __bf16* g, __bf16* l) {
#if HAVE_ASYNC_B128
  __builtin_amdgcn_global_load_async_to_lds_b128((glob_i32x4*)g, (lds_i32x4*)l, 0, 0);
#else
  *(bf16x8*)l = *(const bf16x8*)g;
#endif
}
__device__ __forceinline__ void async_join() {
#if HAVE_ASYNC_B128 && HAVE_WAIT_ASYNC
  __builtin_amdgcn_s_wait_asynccnt(0);
#endif
}

// ---------------- WMMA fragment loaders (contiguous k = 16*(lane/16) + s, same for A and B) ----------------
// A fragment 16x32 bf16, source row-major with leading dim ld (elements).
__device__ __forceinline__ bf16x16 load_frag_a(const __bf16* base, int ld, int lane) {
  int r = lane & 15, g = lane >> 4;
  const __bf16* p = base + r * ld + g * 16;
  bf16x8 lo = *(const bf16x8*)(p);
  bf16x8 hi = *(const bf16x8*)(p + 8);
  bf16x16 f;
#pragma unroll
  for (int i = 0; i < 8; ++i) { f[i] = lo[i]; f[i + 8] = hi[i]; }
  return f;
}
// B fragment 32x16 bf16, source stored N-major [n][k] with leading dim ld.
__device__ __forceinline__ bf16x16 load_frag_b(const __bf16* base, int ld, int lane) {
  int n = lane & 15, g = lane >> 4;
  return *(const bf16x16*)(base + n * ld + g * 16);
}
__device__ __forceinline__ f32x8 wmma_bf16(bf16x16 a, bf16x16 b, f32x8 c) {
  return __builtin_amdgcn_wmma_f32_16x16x32_bf16(false, a, false, b, (short)0, c,
                                                 false, false);
}

// ---------------- block reduction (wave32) ----------------
__device__ __forceinline__ void block_reduce2(float& a, float& b, float* sm, int tid,
                                              int nthreads) {
#pragma unroll
  for (int off = 16; off > 0; off >>= 1) {
    a += __shfl_xor(a, off, 32);
    b += __shfl_xor(b, off, 32);
  }
  int w = tid >> 5;
  if ((tid & 31) == 0) { sm[w] = a; sm[8 + w] = b; }
  __syncthreads();
  if (tid < 32) {
    int nw = (nthreads + 31) >> 5;
    float x = (tid < nw) ? sm[tid] : 0.f;
    float y = (tid < nw) ? sm[8 + tid] : 0.f;
#pragma unroll
    for (int off = 4; off > 0; off >>= 1) { x += __shfl_xor(x, off, 32); y += __shfl_xor(y, off, 32); }
    if (tid == 0) { sm[0] = x; sm[8] = y; }
  }
  __syncthreads();
  a = sm[0]; b = sm[8];
  __syncthreads();
}

// ---------------- elementwise / transform kernels ----------------
__global__ void cvt_bf16_kernel(const float* __restrict__ in, __bf16* __restrict__ out,
                                long n) {
  for (long i = blockIdx.x * (long)blockDim.x + threadIdx.x; i < n;
       i += (long)gridDim.x * blockDim.x)
    out[i] = f2bf(in[i]);
}

// f32 (K x N, row-major) -> bf16 (N x K, row-major), LDS-tiled, z-batched.
__global__ __launch_bounds__(256)
void cvt_transpose_bf16_kernel(const float* __restrict__ in, __bf16* __restrict__ out,
                               int K, int N) {
  __shared__ float tile[32][33];
  long zoff = (long)blockIdx.z * K * N;
  in += zoff;
  out += zoff;
  int k0 = blockIdx.y * 32, n0 = blockIdx.x * 32;
  int tx = threadIdx.x, ty = threadIdx.y;   // (32, 8)
#pragma unroll
  for (int i = ty; i < 32; i += 8)
    tile[i][tx] = in[(long)(k0 + i) * N + (n0 + tx)];
  __syncthreads();
#pragma unroll
  for (int i = ty; i < 32; i += 8)
    out[(long)(n0 + i) * K + (k0 + tx)] = f2bf(tile[tx][i]);
}

__global__ __launch_bounds__(256)
void layernorm_bf16_kernel(const float* __restrict__ X, const float* __restrict__ w,
                           const float* __restrict__ bb, __bf16* __restrict__ out, int D) {
  __shared__ float sm[16];
  long row = blockIdx.x;
  const float* x = X + row * D;
  int tid = threadIdx.x;
  float s = 0.f, s2 = 0.f;
  for (int i = tid; i < D; i += 256) { float v = x[i]; s += v; s2 += v * v; }
  block_reduce2(s, s2, sm, tid, 256);
  float mu = s / (float)D;
  float var = s2 / (float)D - mu * mu;
  float inv = rsqrtf(var + 1e-5f);
  for (int i = tid; i < D; i += 256)
    out[row * D + i] = f2bf((x[i] - mu) * inv * w[i] + bb[i]);
}

// xPos rotary (interleaved pairs) f32 -> bf16.  hd=128 fixed.
__global__ void xpos_tobf16_kernel(const float* __restrict__ In, __bf16* __restrict__ Out,
                                   int S, long total_pairs, int downscale) {
  const int hd = 128, half = 64;
  for (long idx = blockIdx.x * (long)blockDim.x + threadIdx.x; idx < total_pairs;
       idx += (long)gridDim.x * blockDim.x) {
    int p = (int)(idx & (half - 1));
    long rs = idx >> 6;                   // (b*h)*S + s
    int s = (int)(rs % S);
    float base = (2.f * p + 0.4f * hd) / (1.4f * hd);
    float e = (float)s * (1.f / 512.f);
    float scale = powf(base, downscale ? -e : e);
    float invf = powf(10000.f, -(float)p / (float)half);
    float th = (float)s * invf;
    float sn = sinf(th), cs = cosf(th);
    long o = rs * hd + 2 * p;
    float x1 = In[o], x2 = In[o + 1];
    Out[o]     = f2bf((x1 * cs - x2 * sn) * scale);
    Out[o + 1] = f2bf((x2 * cs + x1 * sn) * scale);
  }
}

// per-head GroupNorm over hv + SiLU-style gate, write bf16 (B,S,vd)
__global__ __launch_bounds__(256)
void gn_gate_kernel(const float* __restrict__ Y, const float* __restrict__ G,
                    const float* __restrict__ gw, const float* __restrict__ gb,
                    __bf16* __restrict__ Z, int S, int H) {
  const int hv = 256;
  int s = blockIdx.x, hh = blockIdx.y, b = blockIdx.z;
  int t = threadIdx.x;
  __shared__ float sm[16];
  long yoff = (((long)b * H + hh) * S + s) * hv + t;
  float y = Y[yoff];
  float a = y, a2 = y * y;
  block_reduce2(a, a2, sm, t, 256);
  float mu = a / (float)hv;
  float var = a2 / (float)hv - mu * mu;
  float yn = (y - mu) * rsqrtf(var + 1e-5f);
  int vd = H * hv;
  int j = hh * hv + t;
  long goff = ((long)b * S + s) * (long)vd + j;
  float g = G[goff];
  float sg = 1.f / (1.f + __expf(-g));
  Z[goff] = f2bf(sg * g * (yn * gw[j] + gb[j]));
}

// ---------------- generic bf16 WMMA GEMM ----------------
// C[M,N] = A[M,K] * Bt[N,K]^T (+bias[n]) (+residual); Bt is PRE-TRANSPOSED (N-major).
// mode: 0 = f32 store, 1 = bf16 store, 2 = bf16 gelu(acc+bias), 3 = bf16 transposed
//       (mode 3: Cb[z*sCz + n*M + m], used to emit V^T for the retention kernel)
__global__ __launch_bounds__(128)
void gemm_bf16_kernel(const __bf16* __restrict__ A, const __bf16* __restrict__ Bt,
                      float* __restrict__ Cf, __bf16* __restrict__ Cb,
                      const float* __restrict__ bias, const float* __restrict__ residual,
                      int M, int N, int K, int nH,
                      long sAb, long sAh, long sBh, long sCz, int mode) {
  int z = blockIdx.z;
  int b = z / nH, hh = z % nH;
  A += (long)b * sAb + (long)hh * sAh;
  Bt += (long)hh * sBh;
  long cOff = (long)z * sCz;

  __shared__ alignas(32) __bf16 At[64 * 32];   // [m][k]
  __shared__ alignas(32) __bf16 Bs[64 * 32];   // [n][k]

  int tid = threadIdx.x;
  int lane = tid & 31, wid = tid >> 5;
  int wr = wid >> 1, wc = wid & 1;             // 2x2 wave grid -> 32x32 per wave
  int m0 = blockIdx.y * 64, n0 = blockIdx.x * 64;

  f32x8 acc[2][2] = {};

  for (int k0 = 0; k0 < K; k0 += 32) {
    // A and B tiles: contiguous async 16B chunks (2 + 2 per thread)
#pragma unroll
    for (int c = 0; c < 2; ++c) {
      int idx = tid + 128 * c;                 // 0..255
      int row = idx >> 2, col8 = (idx & 3) * 8;
      copy16_g2lds(A + (long)(m0 + row) * K + k0 + col8, &At[row * 32 + col8]);
      copy16_g2lds(Bt + (long)(n0 + row) * K + k0 + col8, &Bs[row * 32 + col8]);
    }
    async_join();
    __syncthreads();

    if (k0 + 32 < K) {                         // prefetch next K tile
      __builtin_prefetch(A + (long)(m0 + (tid >> 1)) * K + (k0 + 32), 0, 1);
      __builtin_prefetch(Bt + (long)(n0 + (tid >> 1)) * K + (k0 + 32), 0, 1);
    }

#pragma unroll
    for (int mi = 0; mi < 2; ++mi) {
      bf16x16 a = load_frag_a(&At[(wr * 32 + mi * 16) * 32], 32, lane);
#pragma unroll
      for (int ni = 0; ni < 2; ++ni) {
        bf16x16 bfr = load_frag_b(&Bs[(wc * 32 + ni * 16) * 32], 32, lane);
        acc[mi][ni] = wmma_bf16(a, bfr, acc[mi][ni]);
      }
    }
    __syncthreads();
  }

  // epilogue: C layout -> row = i + 8*(lane>=16), col = lane&15
  int hi8 = 8 * (lane >> 4);
  int cl = lane & 15;
#pragma unroll
  for (int mi = 0; mi < 2; ++mi) {
#pragma unroll
    for (int ni = 0; ni < 2; ++ni) {
#pragma unroll
      for (int i = 0; i < 8; ++i) {
        int gr = m0 + wr * 32 + mi * 16 + i + hi8;
        int gc = n0 + wc * 32 + ni * 16 + cl;
        if (gr < M && gc < N) {
          float v = acc[mi][ni][i];
          if (bias) v += bias[gc];
          if (residual) v += residual[cOff + (long)gr * N + gc];
          if (mode == 0)      Cf[cOff + (long)gr * N + gc] = v;
          else if (mode == 1) Cb[cOff + (long)gr * N + gc] = f2bf(v);
          else if (mode == 2) Cb[cOff + (long)gr * N + gc] = f2bf(gelu_exact(v));
          else                Cb[cOff + (long)gc * M + gr] = f2bf(v);   // transposed
        }
      }
    }
  }
}

// ---------------- retention core ----------------
// Y(b,h,s,hv) = sum_t D(s,t) * (Q(s,:)·K(t,:)) * V(t,:)   with D = gamma_h^(s-t), s>=t
// hd=128, hv=256.  VT is pre-transposed: (b,h,hv,S).
// Block: 32 s-rows, 128 threads = 4 waves (2 row x 2 col).
__global__ __launch_bounds__(128)
void retention_kernel(const __bf16* __restrict__ Qb, const __bf16* __restrict__ Kb,
                      const __bf16* __restrict__ VT, float* __restrict__ Y, int S) {
  const int hd = 128, hv = 256;
  int sb = blockIdx.x, hh = blockIdx.y, b = blockIdx.z;
  int H = gridDim.y;
  long base = ((long)b * H + hh) * S;
  const __bf16* Qp = Qb + base * hd;
  const __bf16* Kp = Kb + base * hd;
  const __bf16* VTp = VT + ((long)b * H + hh) * ((long)hv * S);  // [hv][S]
  float* Yp = Y + base * hv;

  __shared__ alignas(32) __bf16 qt[32 * 128];
  __shared__ alignas(32) __bf16 kt[32 * 128];
  __shared__ alignas(32) __bf16 vtT[256 * 32];  // [hv][t]
  __shared__ alignas(32) __bf16 st[32 * 32];    // scores [s][t]

  int tid = threadIdx.x, lane = tid & 31, wid = tid >> 5;
  int wr = wid >> 1, wc = wid & 1;
  int hi8 = 8 * (lane >> 4);
  int cl = lane & 15;

  // decay: gamma_h = 1 - exp(linspace(log(1/32), log(1/512), 8))
  const float l0 = -3.46573590280f, l1 = -6.23832462504f;
  float lg = logf(1.f - __expf(l0 + (float)hh * ((l1 - l0) / 7.f)));

  // Q tile (persistent)
#pragma unroll
  for (int c = 0; c < 4; ++c) {
    int idx = tid + 128 * c;
    int r = idx >> 4, col8 = (idx & 15) * 8;
    copy16_g2lds(Qp + (long)(sb * 32 + r) * hd + col8, &qt[r * 128 + col8]);
  }
  async_join();

  f32x8 acc[8] = {};

  for (int tb = 0; tb <= sb; ++tb) {
    __syncthreads();  // previous iteration's reads done; qt visible on first pass
    // K tile (32 x 128), contiguous async chunks
#pragma unroll
    for (int c = 0; c < 4; ++c) {
      int idx = tid + 128 * c;
      int r = idx >> 4, col8 = (idx & 15) * 8;
      copy16_g2lds(Kp + (long)(tb * 32 + r) * hd + col8, &kt[r * 128 + col8]);
    }
    // V^T tile (256 x 32), contiguous async chunks (pre-transposed in global)
#pragma unroll
    for (int c = 0; c < 8; ++c) {
      int idx = tid + 128 * c;                 // 0..1023
      int cc = idx >> 2, col8 = (idx & 3) * 8;
      copy16_g2lds(VTp + (long)cc * S + tb * 32 + col8, &vtT[cc * 32 + col8]);
    }
    async_join();
    __syncthreads();

    // scores: each wave one 16x16 frag, K=128
    f32x8 sc = {};
#pragma unroll
    for (int kk = 0; kk < 128; kk += 32) {
      bf16x16 a = load_frag_a(&qt[(wr * 16) * 128 + kk], 128, lane);
      bf16x16 bb = load_frag_b(&kt[(wc * 16) * 128 + kk], 128, lane);
      sc = wmma_bf16(a, bb, sc);
    }
    // decay + causal mask, convert to bf16 through LDS (C-layout -> A-layout)
    int sg0 = sb * 32 + wr * 16 + hi8;
    int tg = tb * 32 + wc * 16 + cl;
#pragma unroll
    for (int i = 0; i < 8; ++i) {
      float diff = (float)(sg0 + i - tg);
      float w = (diff >= 0.f) ? __expf(diff * lg) : 0.f;
      st[(wr * 16 + i + hi8) * 32 + wc * 16 + cl] = f2bf(sc[i] * w);
    }
    __syncthreads();

    // Y += scores(16x32) @ V(32x128 per col-wave)
    bf16x16 a = load_frag_a(&st[(wr * 16) * 32], 32, lane);
#pragma unroll
    for (int ni = 0; ni < 8; ++ni) {
      bf16x16 bb = load_frag_b(&vtT[(wc * 128 + ni * 16) * 32], 32, lane);
      acc[ni] = wmma_bf16(a, bb, acc[ni]);
    }
  }

  int rbase = sb * 32 + wr * 16;
#pragma unroll
  for (int ni = 0; ni < 8; ++ni)
#pragma unroll
    for (int i = 0; i < 8; ++i)
      Yp[(long)(rbase + i + hi8) * hv + wc * 128 + ni * 16 + cl] = acc[ni][i];
}

// ---------------- host side ----------------
static inline void gemm(const __bf16* A, const __bf16* Bt, float* Cf, __bf16* Cb,
                        const float* bias, const float* res, int M, int N, int K,
                        int zdim, int nH, long sAb, long sAh, long sBh, long sCz,
                        int mode, hipStream_t s) {
  dim3 g(N / 64, M / 64, zdim);
  gemm_bf16_kernel<<<g, dim3(128), 0, s>>>(A, Bt, Cf, Cb, bias, res, M, N, K, nH, sAb,
                                           sAh, sBh, sCz, mode);
}
static inline void cvt(const float* in, __bf16* out, long n, hipStream_t s) {
  cvt_bf16_kernel<<<dim3(2048), dim3(256), 0, s>>>(in, out, n);
}
// weight convert+transpose: in (z, K, N) f32 -> out (z, N, K) bf16
static inline void cvtT(const float* in, __bf16* out, int K, int N, int z,
                        hipStream_t s) {
  cvt_transpose_bf16_kernel<<<dim3(N / 32, K / 32, z), dim3(32, 8), 0, s>>>(in, out, K,
                                                                            N);
}

extern "C" void kernel_launch(void* const* d_in, const int* in_sizes, int n_in,
                              void* d_out, int out_size, void* d_ws, size_t ws_size,
                              hipStream_t stream) {
  const int B = 2, S = 2048, D = 1024, H = 8, HD = 128, HV = 256, VD = 2048, F = 4096;
  const long BS = (long)B * S;

  const float* Xin = (const float*)d_in[0];
  const float* mem = (const float*)d_in[1];
  const float* lnw[2] = {(const float*)d_in[2], (const float*)d_in[4]};
  const float* lnb[2] = {(const float*)d_in[3], (const float*)d_in[5]};
  const float* Wq_f[2] = {(const float*)d_in[6], (const float*)d_in[13]};
  const float* Wk_f[2] = {(const float*)d_in[7], (const float*)d_in[14]};
  const float* Wv_f[2] = {(const float*)d_in[8], (const float*)d_in[15]};
  const float* Wg_f[2] = {(const float*)d_in[9], (const float*)d_in[16]};
  const float* Wo_f[2] = {(const float*)d_in[10], (const float*)d_in[17]};
  const float* gnw[2] = {(const float*)d_in[11], (const float*)d_in[18]};
  const float* gnb[2] = {(const float*)d_in[12], (const float*)d_in[19]};
  const float* W1 = (const float*)d_in[20];
  const float* b1 = (const float*)d_in[21];
  const float* W2 = (const float*)d_in[22];
  const float* b2 = (const float*)d_in[23];

  float* Xcur = (float*)d_out;

  // workspace carve (bytes, 256-aligned)
  char* w = (char*)d_ws;
  auto carve = [&](size_t bytes) {
    char* p = w;
    w += (bytes + 255) & ~(size_t)255;
    return p;
  };
  const long elQ = (long)H * D * HD;          // 1M
  const long elV = (long)H * D * HV;          // 2M
  const long elG = (long)D * VD;              // 2M
  __bf16* WB = (__bf16*)carve(8ll * 1024 * 1024 * 2);      // stage weights (8M bf16)
  __bf16* wQ = WB, *wK = WB + elQ, *wV = WB + 2 * elQ;     // transposed per head
  __bf16* wG = WB + 2 * elQ + elV, *wO = wG + elG;
  __bf16* w1b = WB;                 // FFN stage reuses region: W1^T (4M) + W2^T (4M)
  __bf16* w2b = WB + (long)D * F;
  __bf16* XnB = (__bf16*)carve(BS * D * 2);
  __bf16* MemB = (__bf16*)carve(BS * D * 2);
  __bf16* QbB = (__bf16*)carve((long)B * H * S * HD * 2);
  __bf16* KbB = (__bf16*)carve((long)B * H * S * HD * 2);
  __bf16* VbT = (__bf16*)carve((long)B * H * S * HV * 2);  // (b,h,hv,S)
  __bf16* ZbB = (__bf16*)carve(BS * VD * 2);
  __bf16* HbB = (__bf16*)carve(BS * F * 2);
  float* Qf = (float*)carve((long)B * H * S * HD * 4);
  float* Kf = (float*)carve((long)B * H * S * HD * 4);
  float* Yf = (float*)carve((long)B * H * S * HV * 4);
  float* Gf = Qf;  // alias: Gf (BS*VD f32) reuses Qf+Kf region after xPos consumes them

  // residual stream init + memory context to bf16
  (void)hipMemcpyAsync(Xcur, Xin, BS * D * sizeof(float), hipMemcpyDeviceToDevice,
                       stream);
  cvt(mem, MemB, BS * D, stream);

  const long pairs = (long)B * H * S * 64;

  for (int st = 0; st < 2; ++st) {
    // weights -> transposed bf16 (N-major for direct async B-tile loads)
    cvtT(Wq_f[st], wQ, D, HD, H, stream);
    cvtT(Wk_f[st], wK, D, HD, H, stream);
    cvtT(Wv_f[st], wV, D, HV, H, stream);
    cvtT(Wg_f[st], wG, D, VD, 1, stream);
    cvtT(Wo_f[st], wO, VD, D, 1, stream);
    // Xq = LN(Xcur)
    layernorm_bf16_kernel<<<dim3((unsigned)BS), dim3(256), 0, stream>>>(
        Xcur, lnw[st], lnb[st], XnB, D);
    const __bf16* Akv = st ? MemB : XnB;  // self-retention vs cross (raw mem)
    // projections (batched over B*H, per-head weight stride)
    gemm(XnB, wQ, Qf, nullptr, nullptr, nullptr, S, HD, D, B * H, H, (long)S * D, 0,
         (long)D * HD, (long)S * HD, 0, stream);
    gemm(Akv, wK, Kf, nullptr, nullptr, nullptr, S, HD, D, B * H, H, (long)S * D, 0,
         (long)D * HD, (long)S * HD, 0, stream);
    // V projection stored TRANSPOSED (mode 3) -> (b,h,hv,S)
    gemm(Akv, wV, nullptr, VbT, nullptr, nullptr, S, HV, D, B * H, H, (long)S * D, 0,
         (long)D * HV, (long)HV * S, 3, stream);
    // xPos rotary -> bf16 Q/K
    xpos_tobf16_kernel<<<dim3(4096), dim3(256), 0, stream>>>(Qf, QbB, S, pairs, 0);
    xpos_tobf16_kernel<<<dim3(4096), dim3(256), 0, stream>>>(Kf, KbB, S, pairs, 1);
    // gate pre-activation G = Xq @ Wg (f32)
    gemm(XnB, wG, Gf, nullptr, nullptr, nullptr, (int)BS, VD, D, 1, 1, 0, 0, 0, 0, 0,
         stream);
    // retention core
    retention_kernel<<<dim3(S / 32, H, B), dim3(128), 0, stream>>>(QbB, KbB, VbT, Yf, S);
    // groupnorm + gate -> Z (bf16)
    gn_gate_kernel<<<dim3(S, H, B), dim3(256), 0, stream>>>(Yf, Gf, gnw[st], gnb[st],
                                                            ZbB, S, H);
    // output projection + residual, in place on Xcur
    gemm(ZbB, wO, Xcur, nullptr, nullptr, Xcur, (int)BS, D, VD, 1, 1, 0, 0, 0, 0, 0,
         stream);
  }

  // FFN
  cvtT(W1, w1b, D, F, 1, stream);
  cvtT(W2, w2b, F, D, 1, stream);
  layernorm_bf16_kernel<<<dim3((unsigned)BS), dim3(256), 0, stream>>>(Xcur, lnw[1],
                                                                      lnb[1], XnB, D);
  gemm(XnB, w1b, nullptr, HbB, b1, nullptr, (int)BS, F, D, 1, 1, 0, 0, 0, 0, 2, stream);
  gemm(HbB, w2b, Xcur, nullptr, b2, Xcur, (int)BS, D, F, 1, 1, 0, 0, 0, 0, 0, stream);
}
